// MultiHead_90374701843215
// MI455X (gfx1250) — matrix operand order repaired and checked by reading, not compiled, not run on hardware
//
#include <hip/hip_runtime.h>
#include <hip/hip_bf16.h>

// ---- problem constants (match reference) ----
constexpr int NUM_HEADS = 16;
constexpr int EMBED_DIM = 1024;
constexpr int HEAD_DIM  = 64;
constexpr int SEQ_LEN   = 2048;
constexpr int BATCH     = 2;
constexpr int HD        = NUM_HEADS * HEAD_DIM; // 1024

typedef __attribute__((ext_vector_type(16))) __bf16        v16bf;
typedef __attribute__((ext_vector_type(8)))  float         v8f;
typedef __attribute__((ext_vector_type(4)))  unsigned int  u32x4;
typedef __attribute__((ext_vector_type(4)))  int           i32x4;
typedef __attribute__((ext_vector_type(8)))  int           i32x8;

union BFrag { v16bf v; u32x4 q[2]; unsigned short u[16]; };
union Pack8 { u32x4 q; unsigned short s[8]; };

#if defined(__gfx1250__) && __has_builtin(__builtin_amdgcn_tensor_load_to_lds)
#define USE_TDM 1
#else
#define USE_TDM 0
#endif

__device__ inline unsigned short f32_to_bf16(float f) {
  __bf16 h = (__bf16)f;                       // native cvt, RNE
  return __builtin_bit_cast(unsigned short, h);
}

__device__ inline unsigned int pack_bf16x2(float lo, float hi) {
  return (unsigned int)f32_to_bf16(lo) | ((unsigned int)f32_to_bf16(hi) << 16);
}

__device__ inline v8f wmma_bf16(v16bf a, v16bf b, v8f c) {
  return __builtin_amdgcn_wmma_f32_16x16x32_bf16(false, a, false, b, (short)0, c,
                                                 false, false);
}

// A-frag 16x32 bf16 from row-major LDS tile: lane half kh in {0,8} covers
// K kh..kh+7 and 16+kh..16+kh+7 -> two contiguous 16B chunks.
template<int PITCH>
__device__ inline v16bf load_a_frag(const unsigned short* lds, int row_base, int col_base) {
  int lane = threadIdx.x & 31;
  int row  = row_base + (lane & 15);
  int kh   = (lane >> 4) * 8;
  BFrag f;
  f.q[0] = *reinterpret_cast<const u32x4*>(&lds[row * PITCH + col_base + kh]);
  f.q[1] = *reinterpret_cast<const u32x4*>(&lds[row * PITCH + col_base + 16 + kh]);
  return f.v;
}

// B-frag 32x16 bf16, LDS storage N-major: B[kk][n] = lds[(n_base+n)*PITCH + k_base+kk].
// Lane half kh in {0,8} covers K kh..kh+7 and 16+kh..16+kh+7 -> two 16B chunks.
template<int PITCH>
__device__ inline v16bf load_b_frag_nmaj(const unsigned short* lds, int n_base, int k_base) {
  int lane = threadIdx.x & 31;
  int n    = lane & 15;
  int kh16 = (lane >> 4) * 8;
  BFrag f;
  const unsigned short* p = &lds[(n_base + n) * PITCH + k_base + kh16];
  f.q[0] = *reinterpret_cast<const u32x4*>(p);
  f.q[1] = *reinterpret_cast<const u32x4*>(p + 16);
  return f.v;
}

#define V8F_ZERO {0.f,0.f,0.f,0.f,0.f,0.f,0.f,0.f}

// Transposed bf16 staging of a 64x64 fp32 weight tile: WT[d][e], pitch 72.
__device__ inline void fill_wt_tile64(unsigned short* WT, const float* WH, int e0, int t) {
#pragma unroll
  for (int i = 0; i < 4; ++i) {
    int slot = i * 256 + t;                // 1024 float4 groups
    int r = slot >> 4, cg = slot & 15;     // r: e row 0..63, cg: 4-wide d group
    float4 v = *reinterpret_cast<const float4*>(&WH[(size_t)(e0 + r) * HEAD_DIM + cg * 4]);
    int c = cg * 4;
    WT[(c + 0) * 72 + r] = f32_to_bf16(v.x);
    WT[(c + 1) * 72 + r] = f32_to_bf16(v.y);
    WT[(c + 2) * 72 + r] = f32_to_bf16(v.z);
    WT[(c + 3) * 72 + r] = f32_to_bf16(v.w);
  }
}

#if USE_TDM
// TDM: DMA a 2D bf16 tile (tile_d0 contiguous elems x tile_d1 rows, row stride
// stride0 elems) into LDS, padding pad_amount(+1) dwords after every
// 2^(pad_interval+1) dwords.
__device__ inline void tdm_load_2d(unsigned lds_addr, const void* gptr,
                                   unsigned tile_d0, unsigned tile_d1,
                                   unsigned stride0, unsigned tensor_d0,
                                   unsigned tensor_d1,
                                   unsigned pad_interval, unsigned pad_amount) {
  unsigned long long ga = (unsigned long long)(uintptr_t)gptr;
  u32x4 g0;
  g0[0] = 1u;                                   // count=1, user mode, no gather
  g0[1] = lds_addr;                             // LDS byte address
  g0[2] = (unsigned)ga;                         // global_addr[31:0]
  g0[3] = (unsigned)((ga >> 32) & 0x1FFFFFFu)   // global_addr[56:32]
        | (2u << 30);                           // type=2 (image)
  i32x8 g1;
  g1[0] = (int)((1u << 16)                      // data_size = 2 bytes
              | (1u << 20)                      // pad_enable
              | (pad_interval << 22)
              | (pad_amount << 25));
  g1[1] = (int)(tensor_d0 << 16);                         // tensor_dim0 lo16
  g1[2] = (int)((tensor_d0 >> 16) | (tensor_d1 << 16));   // dim0 hi | dim1 lo
  g1[3] = (int)((tensor_d1 >> 16) | (tile_d0 << 16));     // dim1 hi | tile_dim0
  g1[4] = (int)(tile_d1 & 0xFFFFu);                       // tile_dim1, tile_dim2=0
  g1[5] = (int)stride0;                                   // tensor_dim0_stride lo32
  g1[6] = 0;
  g1[7] = 0;
  i32x4 gz = {0, 0, 0, 0};
#if __clang_major__ >= 23
  i32x8 gz8 = {0, 0, 0, 0, 0, 0, 0, 0};
  __builtin_amdgcn_tensor_load_to_lds(g0, g1, gz, gz, gz8, 0);
#else
  __builtin_amdgcn_tensor_load_to_lds(g0, g1, gz, gz, 0);
#endif
}
#endif

// ------------------------------------------------------------------
// Kernel 1: Q/K/V projections. grid (64, 16). K-step 64 per stage.
// Q,K -> bf16 [B,H,S,D]; V -> bf16 [B,H,D,S] (pre-transposed for TDM).
// ------------------------------------------------------------------
__global__ void __launch_bounds__(256) qkv_proj_kernel(
    const float* __restrict__ x,
    const float* __restrict__ Wq, const float* __restrict__ Wk,
    const float* __restrict__ Wv,
    unsigned short* __restrict__ Qb, unsigned short* __restrict__ Kb,
    unsigned short* __restrict__ Vb)
{
  __shared__ __align__(16) unsigned short Xs[64 * 72];    // x tile 64x64, pitch 72
  __shared__ __align__(16) unsigned short WT0[64 * 72];   // W tiles transposed [d][e]
  __shared__ __align__(16) unsigned short WT1[64 * 72];
  __shared__ __align__(16) unsigned short WT2[64 * 72];

  const int rt = blockIdx.x;
  const int h  = blockIdx.y;
  const int row0 = rt * 64;
  const int b  = row0 / SEQ_LEN;
  const int s0 = row0 - b * SEQ_LEN;
  const int t  = threadIdx.x;
  const int w  = t >> 5;
  const int wm = w & 3;
  const int wn = w >> 2;

  const float* WqH = Wq + (size_t)h * EMBED_DIM * HEAD_DIM;
  const float* WkH = Wk + (size_t)h * EMBED_DIM * HEAD_DIM;
  const float* WvH = Wv + (size_t)h * EMBED_DIM * HEAD_DIM;

  v8f aQ0 = V8F_ZERO, aQ1 = V8F_ZERO;
  v8f aK0 = V8F_ZERO, aK1 = V8F_ZERO;
  v8f aV0 = V8F_ZERO, aV1 = V8F_ZERO;

  for (int e0 = 0; e0 < EMBED_DIM; e0 += 64) {
    __syncthreads();
#pragma unroll
    for (int i = 0; i < 4; ++i) {            // X tile 64x64 fp32 -> bf16
      int slot = i * 256 + t;                // 1024 float4 groups
      int r = slot >> 4, cg = slot & 15;
      float4 v = *reinterpret_cast<const float4*>(
          &x[(size_t)(b * SEQ_LEN + s0 + r) * EMBED_DIM + e0 + cg * 4]);
      unsigned int* dst = reinterpret_cast<unsigned int*>(&Xs[r * 72 + cg * 4]);
      dst[0] = pack_bf16x2(v.x, v.y);
      dst[1] = pack_bf16x2(v.z, v.w);
    }
    fill_wt_tile64(WT0, WqH, e0, t);
    fill_wt_tile64(WT1, WkH, e0, t);
    fill_wt_tile64(WT2, WvH, e0, t);
    __syncthreads();

    v16bf a0 = load_a_frag<72>(Xs, wm * 16, 0);
    v16bf a1 = load_a_frag<72>(Xs, wm * 16, 32);
    aQ0 = wmma_bf16(a0, load_b_frag_nmaj<72>(WT0, wn * 32,      0),  aQ0);
    aQ0 = wmma_bf16(a1, load_b_frag_nmaj<72>(WT0, wn * 32,      32), aQ0);
    aQ1 = wmma_bf16(a0, load_b_frag_nmaj<72>(WT0, wn * 32 + 16, 0),  aQ1);
    aQ1 = wmma_bf16(a1, load_b_frag_nmaj<72>(WT0, wn * 32 + 16, 32), aQ1);
    aK0 = wmma_bf16(a0, load_b_frag_nmaj<72>(WT1, wn * 32,      0),  aK0);
    aK0 = wmma_bf16(a1, load_b_frag_nmaj<72>(WT1, wn * 32,      32), aK0);
    aK1 = wmma_bf16(a0, load_b_frag_nmaj<72>(WT1, wn * 32 + 16, 0),  aK1);
    aK1 = wmma_bf16(a1, load_b_frag_nmaj<72>(WT1, wn * 32 + 16, 32), aK1);
    aV0 = wmma_bf16(a0, load_b_frag_nmaj<72>(WT2, wn * 32,      0),  aV0);
    aV0 = wmma_bf16(a1, load_b_frag_nmaj<72>(WT2, wn * 32,      32), aV0);
    aV1 = wmma_bf16(a0, load_b_frag_nmaj<72>(WT2, wn * 32 + 16, 0),  aV1);
    aV1 = wmma_bf16(a1, load_b_frag_nmaj<72>(WT2, wn * 32 + 16, 32), aV1);
  }

  const int lane  = t & 31;
  const int ncol  = lane & 15;
  const int mbase = (lane >> 4) * 8;
  const int c0    = wn * 32 + ncol;
  const int srow0 = s0 + wm * 16 + mbase;
#pragma unroll
  for (int j = 0; j < 8; ++j) {              // Q,K: [B,H,S,D]
    int srow = srow0 + j;
    size_t base = (((size_t)b * NUM_HEADS + h) * SEQ_LEN + srow) * HEAD_DIM;
    Qb[base + c0]      = f32_to_bf16(aQ0[j] * 0.125f);
    Qb[base + c0 + 16] = f32_to_bf16(aQ1[j] * 0.125f);
    Kb[base + c0]      = f32_to_bf16(aK0[j]);
    Kb[base + c0 + 16] = f32_to_bf16(aK1[j]);
  }
  {                                          // V: transposed [B,H,D,S], b128 stores
    Pack8 v0, v1;
#pragma unroll
    for (int j = 0; j < 8; ++j) {
      v0.s[j] = f32_to_bf16(aV0[j]);
      v1.s[j] = f32_to_bf16(aV1[j]);
    }
    size_t vbase = ((size_t)b * NUM_HEADS + h) * HEAD_DIM;
    *reinterpret_cast<u32x4*>(&Vb[(vbase + c0)      * SEQ_LEN + srow0]) = v0.q;
    *reinterpret_cast<u32x4*>(&Vb[(vbase + c0 + 16) * SEQ_LEN + srow0]) = v1.q;
  }
}

// ------------------------------------------------------------------
// Kernel 2: causal flash attention. grid (S/128, B*H).
// K and V tiles fetched by the Tensor Data Mover.
// ------------------------------------------------------------------
__global__ void __launch_bounds__(256) attn_kernel(
    const unsigned short* __restrict__ Qb, const unsigned short* __restrict__ Kb,
    const unsigned short* __restrict__ Vb, unsigned short* __restrict__ Cb)
{
  __shared__ __align__(16) unsigned short Ks[32 * 72];      // K tile [sk][d], pitch 72
  __shared__ __align__(16) unsigned short VsT[64 * 40];     // V tile [d][sk], pitch 40
  __shared__ __align__(16) unsigned short Ps[8][16 * 40];   // per-wave P scratch

  const int qt = blockIdx.x;
  const int bh = blockIdx.y;
  const int b = bh >> 4, h = bh & 15;
  const int q0 = qt * 128;
  const int t = threadIdx.x, w = t >> 5, lane = t & 31;
  const int qw = q0 + w * 16;

  size_t headoff = ((size_t)b * NUM_HEADS + h) * SEQ_LEN * HEAD_DIM;
  const unsigned short* Qh  = Qb + headoff;
  const unsigned short* Kh  = Kb + headoff;
  const unsigned short* Vth = Vb + headoff;    // [D][S] layout

  // preload Q A-frags (D=64: two K-steps) straight from global, b128 loads
  BFrag qa[2];
  {
    int row = qw + (lane & 15);
    int kh  = (lane >> 4) * 8;
    const unsigned short* qp = Qh + (size_t)row * HEAD_DIM;
#pragma unroll
    for (int ks = 0; ks < 2; ++ks) {
      qa[ks].q[0] = *reinterpret_cast<const u32x4*>(qp + ks * 32 + kh);
      qa[ks].q[1] = *reinterpret_cast<const u32x4*>(qp + ks * 32 + 16 + kh);
    }
  }

  v8f accO[4] = {V8F_ZERO, V8F_ZERO, V8F_ZERO, V8F_ZERO};
  float mrun[8], lrun[8];
#pragma unroll
  for (int j = 0; j < 8; ++j) { mrun[j] = -1e30f; lrun[j] = 0.f; }

  const int mbase = (lane >> 4) * 8;
  const int ncol  = lane & 15;
  const int ktiles = q0 / 32 + 4;

#if USE_TDM
  const unsigned KsAddr = (unsigned)(uintptr_t)(&Ks[0]);
  const unsigned VsAddr = (unsigned)(uintptr_t)(&VsT[0]);
#endif

  for (int kt = 0; kt < ktiles; ++kt) {
    const int sk0 = kt * 32;
    __syncthreads();                            // prev-iter reads done
#if USE_TDM
    // K: 64x32 tile of [S,D] tensor, pad 4dw per 32dw -> pitch 144B (72 ushorts)
    if (w == 0) tdm_load_2d(KsAddr, Kh + (size_t)sk0 * HEAD_DIM,
                            64, 32, 64, 64, SEQ_LEN, 4, 3);
    // V: 32x64 tile of [D,S] tensor, pad 4dw per 16dw -> pitch 80B (40 ushorts)
    if (w == 1) tdm_load_2d(VsAddr, Vth + sk0,
                            32, 64, SEQ_LEN, SEQ_LEN, 64, 3, 3);
    if (w < 2) __builtin_amdgcn_s_wait_tensorcnt(0);
#else
    {
      int r = t >> 3, cg = t & 7;               // K tile: 256 slots of 8 elems
      *reinterpret_cast<u32x4*>(&Ks[r * 72 + cg * 8]) =
          *reinterpret_cast<const u32x4*>(&Kh[(size_t)(sk0 + r) * HEAD_DIM + cg * 8]);
    }
    {
      int d = t >> 2, cg = t & 3;               // V tile: straight copy of [D,S] rows
      *reinterpret_cast<u32x4*>(&VsT[d * 40 + cg * 8]) =
          *reinterpret_cast<const u32x4*>(&Vth[(size_t)d * SEQ_LEN + sk0 + cg * 8]);
    }
#endif
    __syncthreads();

    // S = Q*K^T : M=16, N=32 (2 tiles), K=64 (2 steps). K tile is N-major.
    v8f sc0 = V8F_ZERO, sc1 = V8F_ZERO;
    sc0 = wmma_bf16(qa[0].v, load_b_frag_nmaj<72>(Ks, 0,  0),  sc0);
    sc0 = wmma_bf16(qa[1].v, load_b_frag_nmaj<72>(Ks, 0,  32), sc0);
    sc1 = wmma_bf16(qa[0].v, load_b_frag_nmaj<72>(Ks, 16, 0),  sc1);
    sc1 = wmma_bf16(qa[1].v, load_b_frag_nmaj<72>(Ks, 16, 32), sc1);

    // causal mask + online softmax
    unsigned short* Pw = Ps[w];
#pragma unroll
    for (int j = 0; j < 8; ++j) {
      int qrow = qw + mbase + j;
      float s0v = sc0[j]; if (sk0 + ncol      > qrow) s0v = -1e30f;
      float s1v = sc1[j]; if (sk0 + 16 + ncol > qrow) s1v = -1e30f;
      float mx = fmaxf(s0v, s1v);
#pragma unroll
      for (int off = 8; off; off >>= 1) mx = fmaxf(mx, __shfl_xor(mx, off, 16));
      float mnew  = fmaxf(mrun[j], mx);
      float scale = __expf(mrun[j] - mnew);
      float p0 = __expf(s0v - mnew);
      float p1 = __expf(s1v - mnew);
      float ps = p0 + p1;
#pragma unroll
      for (int off = 8; off; off >>= 1) ps += __shfl_xor(ps, off, 16);
      lrun[j] = lrun[j] * scale + ps;
      mrun[j] = mnew;
      accO[0][j] *= scale; accO[1][j] *= scale;
      accO[2][j] *= scale; accO[3][j] *= scale;
      Pw[(mbase + j) * 40 + ncol]      = f32_to_bf16(p0);
      Pw[(mbase + j) * 40 + 16 + ncol] = f32_to_bf16(p1);
    }
    __syncthreads();                            // P visible for A-frag reload

    // O += P*V : M=16, N=64 (4 tiles), K=32. V is N-major in LDS.
    v16bf pa = load_a_frag<40>(Pw, 0, 0);
#pragma unroll
    for (int nt = 0; nt < 4; ++nt)
      accO[nt] = wmma_bf16(pa, load_b_frag_nmaj<40>(VsT, nt * 16, 0), accO[nt]);
  }

  // epilogue: O /= l, write concat [B,S,H*D] bf16
#pragma unroll
  for (int j = 0; j < 8; ++j) {
    float inv = 1.0f / lrun[j];
    int srow = qw + mbase + j;
    size_t base = ((size_t)b * SEQ_LEN + srow) * HD + h * HEAD_DIM;
#pragma unroll
    for (int nt = 0; nt < 4; ++nt)
      Cb[base + nt * 16 + ncol] = f32_to_bf16(accO[nt][j] * inv);
  }
}

// ------------------------------------------------------------------
// Kernel 3: output projection [B*S,1024] x [1024,1024] + bias -> fp32
// K-step 64; activation tile via TDM.
// ------------------------------------------------------------------
__global__ void __launch_bounds__(256) out_proj_kernel(
    const unsigned short* __restrict__ Cb, const float* __restrict__ Wo,
    const float* __restrict__ bo, float* __restrict__ out)
{
  __shared__ __align__(16) unsigned short Cs[64 * 72];     // concat tile 64x64, pitch 72
  __shared__ __align__(16) unsigned short WoT[64 * 72];    // Wo tile transposed [n][k]

  const int rt = blockIdx.x, ct = blockIdx.y;
  const int row0 = rt * 64, col0 = ct * 64;
  const int t = threadIdx.x, w = t >> 5, lane = t & 31;
  const int wm = w & 3, wn = w >> 2;

  v8f a0 = V8F_ZERO, a1 = V8F_ZERO;

#if USE_TDM
  const unsigned CsAddr = (unsigned)(uintptr_t)(&Cs[0]);
#endif

  for (int k0 = 0; k0 < HD; k0 += 64) {
    __syncthreads();
#if USE_TDM
    // Cs: 64x64 tile of [4096,1024] bf16 tensor, pad 4dw per 32dw -> pitch 72
    if (w == 0) tdm_load_2d(CsAddr, Cb + (size_t)row0 * HD + k0,
                            64, 64, HD, HD, BATCH * SEQ_LEN, 4, 3);
#else
#pragma unroll
    for (int i = 0; i < 2; ++i) {              // 512 slots of 8 elems
      int slot = i * 256 + t;
      int r = slot >> 3, cg = slot & 7;
      *reinterpret_cast<u32x4*>(&Cs[r * 72 + cg * 8]) =
          *reinterpret_cast<const u32x4*>(&Cb[(size_t)(row0 + r) * HD + k0 + cg * 8]);
    }
#endif
#pragma unroll
    for (int i = 0; i < 4; ++i) {              // WoT: fp32 -> bf16 transposed
      int slot = i * 256 + t;
      int r = slot >> 4, cg = slot & 15;
      float4 v = *reinterpret_cast<const float4*>(
          &Wo[(size_t)(k0 + r) * EMBED_DIM + col0 + cg * 4]);
      int c = cg * 4;
      WoT[(c + 0) * 72 + r] = f32_to_bf16(v.x);
      WoT[(c + 1) * 72 + r] = f32_to_bf16(v.y);
      WoT[(c + 2) * 72 + r] = f32_to_bf16(v.z);
      WoT[(c + 3) * 72 + r] = f32_to_bf16(v.w);
    }
#if USE_TDM
    if (w == 0) __builtin_amdgcn_s_wait_tensorcnt(0);
#endif
    __syncthreads();
    v16bf a0f = load_a_frag<72>(Cs, wm * 16, 0);
    v16bf a1f = load_a_frag<72>(Cs, wm * 16, 32);
    a0 = wmma_bf16(a0f, load_b_frag_nmaj<72>(WoT, wn * 32,      0),  a0);
    a0 = wmma_bf16(a1f, load_b_frag_nmaj<72>(WoT, wn * 32,      32), a0);
    a1 = wmma_bf16(a0f, load_b_frag_nmaj<72>(WoT, wn * 32 + 16, 0),  a1);
    a1 = wmma_bf16(a1f, load_b_frag_nmaj<72>(WoT, wn * 32 + 16, 32), a1);
  }

  const int ncol = lane & 15, mbase = (lane >> 4) * 8;
#pragma unroll
  for (int j = 0; j < 8; ++j) {
    int r = row0 + wm * 16 + mbase + j;
    int c = col0 + wn * 32 + ncol;
    out[(size_t)r * EMBED_DIM + c]      = a0[j] + bo[c];
    out[(size_t)r * EMBED_DIM + c + 16] = a1[j] + bo[c + 16];
  }
}

// ------------------------------------------------------------------
extern "C" void kernel_launch(void* const* d_in, const int* in_sizes, int n_in,
                              void* d_out, int out_size, void* d_ws, size_t ws_size,
                              hipStream_t stream) {
  const float* x  = (const float*)d_in[0];
  const float* Wq = (const float*)d_in[1];
  const float* Wk = (const float*)d_in[2];
  const float* Wv = (const float*)d_in[3];
  const float* Wo = (const float*)d_in[4];
  const float* bo = (const float*)d_in[5];
  float* out = (float*)d_out;

  const size_t qkv = (size_t)BATCH * NUM_HEADS * SEQ_LEN * HEAD_DIM; // 4M elems
  unsigned short* Qb = (unsigned short*)d_ws;      // [B,H,S,D]
  unsigned short* Kb = Qb + qkv;                   // [B,H,S,D]
  unsigned short* Vb = Kb + qkv;                   // [B,H,D,S] (transposed)
  unsigned short* Cb = Vb + qkv;                   // [B,S,H*D]

  qkv_proj_kernel<<<dim3((BATCH * SEQ_LEN) / 64, NUM_HEADS), 256, 0, stream>>>(
      x, Wq, Wk, Wv, Qb, Kb, Vb);
  attn_kernel<<<dim3(SEQ_LEN / 128, BATCH * NUM_HEADS), 256, 0, stream>>>(
      Qb, Kb, Vb, Cb);
  out_proj_kernel<<<dim3((BATCH * SEQ_LEN) / 64, EMBED_DIM / 64), 256, 0, stream>>>(
      Cb, Wo, bo, out);
}
